// CrossAttention_30279519436903
// MI455X (gfx1250) — matrix operand order repaired and checked by reading, not compiled, not run on hardware
//
#include <hip/hip_runtime.h>
#include <hip/hip_bf16.h>

typedef __bf16 bf16_t;
typedef __attribute__((ext_vector_type(16))) __bf16 v16bf;
typedef __attribute__((ext_vector_type(8)))  float  v8f;

union FragBF {            // 16 bf16 operand fragment (A or B) for wmma 16x16x32
    v16bf v;
    uint4 u[2];
};
union AccF {              // 16x16 fp32 C/D tile: 8 VGPRs
    v8f   v;
    float f[8];
};

#define D_MODEL 1024
#define LQ      1024
#define LKV     2048
#define BATCH   4
#define NHEAD   16
#define HDIM    64

// ---------------------------------------------------------------------------
// CDNA5 async global->LDS copy (16B per lane), tracked by ASYNCcnt.
// LDS dst operand = low 32 bits of the flat address.
// ---------------------------------------------------------------------------
__device__ __forceinline__ void async_copy_b128(void* lds_ptr, const void* gptr) {
    uint32_t l = (uint32_t)(uintptr_t)lds_ptr;
    asm volatile("global_load_async_to_lds_b128 %0, %1, off"
                 :: "v"(l), "v"(gptr) : "memory");
}
__device__ __forceinline__ void wait_async0() {
    asm volatile("s_wait_asynccnt 0" ::: "memory");
}

// ---------------------------------------------------------------------------
// fp32 -> bf16 conversion (4 elements / thread)
// ---------------------------------------------------------------------------
__global__ void cvt_f32_bf16(const float* __restrict__ src, bf16_t* __restrict__ dst, int n4) {
    int i = blockIdx.x * blockDim.x + threadIdx.x;
    if (i >= n4) return;
    float4 f = reinterpret_cast<const float4*>(src)[i];
    union { bf16_t h[4]; uint2 u; } o;
    o.h[0] = (bf16_t)f.x; o.h[1] = (bf16_t)f.y; o.h[2] = (bf16_t)f.z; o.h[3] = (bf16_t)f.w;
    reinterpret_cast<uint2*>(dst)[i] = o.u;
}

// ---------------------------------------------------------------------------
// bf16 GEMM:  C[M,N] = A[M,K] * W[N,K]^T + bias[N], row-major K-contiguous.
// Block = 256 thr = 8 waves; wave tile = 32x64; WG tile = 128x128.
// mode 0: bf16 row-major out, mode 1: f32 row-major out,
// mode 2: bf16 "Vt" store -> Vt[((b*16+h)*64+d)*2048 + kv]
// ---------------------------------------------------------------------------
__global__ __launch_bounds__(256)
void gemm_bf16_nt(const bf16_t* __restrict__ A, const bf16_t* __restrict__ W,
                  const float* __restrict__ bias, void* __restrict__ out,
                  int M, int N, int K, int mode, float outScale) {
    const int lane   = threadIdx.x & 31;
    const int w      = threadIdx.x >> 5;
    const int laneLo = lane & 15;
    const bool hi    = lane >= 16;

    const int m0 = blockIdx.y * 128 + (w & 3) * 32;
    const int n0 = blockIdx.x * 128 + (w >> 2) * 64;

    AccF acc[2][4];
#pragma unroll
    for (int i = 0; i < 2; ++i)
#pragma unroll
        for (int j = 0; j < 4; ++j) acc[i][j].v = (v8f)0.0f;

    for (int k0 = 0; k0 < K; k0 += 32) {
        FragBF a[2], b[4];
#pragma unroll
        for (int am = 0; am < 2; ++am) {
            const bf16_t* p = A + (size_t)(m0 + am * 16 + laneLo) * K + k0;
            a[am].u[0] = *reinterpret_cast<const uint4*>(p + (hi ? 8 : 0));
            a[am].u[1] = *reinterpret_cast<const uint4*>(p + 16 + (hi ? 8 : 0));
        }
#pragma unroll
        for (int bn = 0; bn < 4; ++bn) {
            const bf16_t* p = W + (size_t)(n0 + bn * 16 + laneLo) * K + k0;
            b[bn].u[0] = *reinterpret_cast<const uint4*>(p + (hi ? 8 : 0));
            b[bn].u[1] = *reinterpret_cast<const uint4*>(p + 16 + (hi ? 8 : 0));
        }
        if (k0 + 32 < K) {  // WGP-scope prefetch of next A k-slab
            __builtin_prefetch(A + (size_t)(m0 + laneLo) * K + k0 + 32, 0, 3);
        }
#pragma unroll
        for (int am = 0; am < 2; ++am)
#pragma unroll
            for (int bn = 0; bn < 4; ++bn)
                acc[am][bn].v = __builtin_amdgcn_wmma_f32_16x16x32_bf16(
                    false, a[am].v, false, b[bn].v, (short)0, acc[am][bn].v, false, false);
    }

#pragma unroll
    for (int bn = 0; bn < 4; ++bn) {
        const int col = n0 + bn * 16 + laneLo;
        const float bc = bias[col];
#pragma unroll
        for (int am = 0; am < 2; ++am) {
#pragma unroll
            for (int r = 0; r < 8; ++r) {
                const int row = m0 + am * 16 + r + (hi ? 8 : 0);
                const float v = (acc[am][bn].f[r] + bc) * outScale;
                if (mode == 0) {
                    reinterpret_cast<bf16_t*>(out)[(size_t)row * N + col] = (bf16_t)v;
                } else if (mode == 1) {
                    reinterpret_cast<float*>(out)[(size_t)row * N + col] = v;
                } else {  // Vt store: row = b*2048+kv, col = h*64+d
                    const int bb = row >> 11, kv = row & 2047;
                    const int hh = col >> 6,  dd = col & 63;
                    reinterpret_cast<bf16_t*>(out)[(((size_t)(bb * 16 + hh) * 64 + dd) << 11) + kv] =
                        (bf16_t)v;
                }
            }
        }
    }
}

// ---------------------------------------------------------------------------
// Flash attention. One WG = one (b,h) and 128 consecutive q rows (8 waves x
// 16-row q block). K/V chunks (32 kv cols) staged to LDS with double-buffered
// global_load_async_to_lds_b128 and shared by all 8 waves.
// ---------------------------------------------------------------------------
__global__ __launch_bounds__(256)
void attn_flash(const bf16_t* __restrict__ Q, const bf16_t* __restrict__ Kw,
                const bf16_t* __restrict__ Vt, const int* __restrict__ mask,
                bf16_t* __restrict__ O) {
    __shared__ __align__(16) bf16_t lk[2][32 * 64];      // K chunk: kv x d
    __shared__ __align__(16) bf16_t lv[2][64 * 32];      // V chunk: d x kv (from Vt)
    __shared__ __align__(16) bf16_t pbuf[8 * 16 * 32];   // P relay, 1KB/wave

    const int tid    = threadIdx.x;
    const int lane   = tid & 31;
    const int w      = tid >> 5;
    const int laneLo = lane & 15;
    const bool hi    = lane >= 16;

    const int wg   = blockIdx.x;          // 0..511
    const int b    = wg >> 7;             // 128 WGs per batch (16 h * 8 qtiles)
    const int h    = (wg >> 3) & 15;
    const int qb   = (wg & 7) * 8 + w;    // wave's 16-row q block (0..63)

    const bf16_t* Qb = Q  + (size_t)(b * LQ + qb * 16) * D_MODEL + h * HDIM;
    const bf16_t* Kb = Kw + (size_t)(b * LKV) * D_MODEL + h * HDIM;
    const bf16_t* Vb = Vt + (size_t)((b * NHEAD + h) * HDIM) * LKV;
    const int*    mb = mask + b * LKV;
    bf16_t*       pl = pbuf + w * (16 * 32);

    // Per-thread staging slots: K 32x64 (one b128 each), V 64x32 (one b128 each)
    const int krow = tid >> 3, kcol = (tid & 7) * 8;
    const int vrow = tid >> 2, vcol = (tid & 3) * 8;
    const bf16_t* kg = Kb + (size_t)krow * D_MODEL + kcol;
    const bf16_t* vg = Vb + (size_t)vrow * LKV + vcol;
    const int kslot = krow * 64 + kcol;
    const int vslot = vrow * 32 + vcol;

    // Q A-fragments for K-dims 0..31 and 32..63
    FragBF qa[2];
#pragma unroll
    for (int kk = 0; kk < 2; ++kk) {
        const bf16_t* p = Qb + (size_t)laneLo * D_MODEL + kk * 32;
        qa[kk].u[0] = *reinterpret_cast<const uint4*>(p + (hi ? 8 : 0));
        qa[kk].u[1] = *reinterpret_cast<const uint4*>(p + 16 + (hi ? 8 : 0));
    }

    AccF o[4];
#pragma unroll
    for (int t = 0; t < 4; ++t) o[t].v = (v8f)0.0f;
    float mprev[8], lsum[8];
#pragma unroll
    for (int r = 0; r < 8; ++r) { mprev[r] = -3.0e38f; lsum[r] = 0.0f; }

    // Prologue: stage kv chunk 0 into buffer 0
    async_copy_b128(&lk[0][kslot], kg);
    async_copy_b128(&lv[0][vslot], vg);

    for (int kv0 = 0, it = 0; kv0 < LKV; kv0 += 32, ++it) {
        const int cur = it & 1, nxt = cur ^ 1;

        wait_async0();        // this wave's staged data is in LDS
        __syncthreads();      // ... and everyone else's too

        if (kv0 + 32 < LKV) { // overlap next chunk's DMA with this chunk's math
            async_copy_b128(&lk[nxt][kslot], kg + (size_t)(kv0 + 32) * D_MODEL);
            async_copy_b128(&lv[nxt][vslot], vg + (kv0 + 32));
        }

        // ---- S = Q * K^T (16 x 32), two 16x16 C tiles, K frags from LDS ----
        AccF s[2];
#pragma unroll
        for (int t = 0; t < 2; ++t) {
            s[t].v = (v8f)0.0f;
            const bf16_t* kp = &lk[cur][(t * 16 + laneLo) * 64];
#pragma unroll
            for (int kk = 0; kk < 2; ++kk) {
                FragBF kf;
                const bf16_t* p = kp + kk * 32;
                kf.u[0] = *reinterpret_cast<const uint4*>(p + (hi ? 8 : 0));
                kf.u[1] = *reinterpret_cast<const uint4*>(p + 16 + (hi ? 8 : 0));
                s[t].v = __builtin_amdgcn_wmma_f32_16x16x32_bf16(
                    false, qa[kk].v, false, kf.v, (short)0, s[t].v, false, false);
            }
        }

        const int mk0 = mb[kv0 + laneLo];
        const int mk1 = mb[kv0 + 16 + laneLo];

        // ---- online softmax per row; write P (bf16) into wave-private LDS ----
#pragma unroll
        for (int r = 0; r < 8; ++r) {
            float s0 = mk0 ? s[0].f[r] : -3.0e38f;
            float s1 = mk1 ? s[1].f[r] : -3.0e38f;
            float vmax = fmaxf(s0, s1);
#pragma unroll
            for (int off = 1; off < 16; off <<= 1)
                vmax = fmaxf(vmax, __shfl_xor(vmax, off, 32));
            const float mnew  = fmaxf(mprev[r], vmax);
            const float scale = __expf(mprev[r] - mnew);
            const float p0 = __expf(s0 - mnew);
            const float p1 = __expf(s1 - mnew);
            float psum = p0 + p1;
#pragma unroll
            for (int off = 1; off < 16; off <<= 1)
                psum += __shfl_xor(psum, off, 32);
            lsum[r]  = lsum[r] * scale + psum;
            mprev[r] = mnew;
#pragma unroll
            for (int t = 0; t < 4; ++t) o[t].f[r] *= scale;
            const int prow = r + (hi ? 8 : 0);
            pl[prow * 32 + laneLo]      = (bf16_t)p0;
            pl[prow * 32 + 16 + laneLo] = (bf16_t)p1;
        }
        asm volatile("s_wait_dscnt 0" ::: "memory");   // wave-local LDS RAW

        // ---- reload P as an A fragment (16 x 32) ----
        FragBF pf;
        {
            const bf16_t* pp = pl + laneLo * 32;
            pf.u[0] = *reinterpret_cast<const uint4*>(pp + (hi ? 8 : 0));
            pf.u[1] = *reinterpret_cast<const uint4*>(pp + 16 + (hi ? 8 : 0));
        }
        // ---- O += P * V, V frags from LDS ----
#pragma unroll
        for (int t = 0; t < 4; ++t) {
            FragBF vf;
            const bf16_t* vp = &lv[cur][(t * 16 + laneLo) * 32];
            vf.u[0] = *reinterpret_cast<const uint4*>(vp + (hi ? 8 : 0));
            vf.u[1] = *reinterpret_cast<const uint4*>(vp + 16 + (hi ? 8 : 0));
            o[t].v = __builtin_amdgcn_wmma_f32_16x16x32_bf16(
                false, pf.v, false, vf.v, (short)0, o[t].v, false, false);
        }
    }

    // ---- normalize and store O (bf16, [B*Lq, D], col = h*64+d) ----
    bf16_t* Ob = O + (size_t)(b * LQ + qb * 16) * D_MODEL + h * HDIM;
#pragma unroll
    for (int r = 0; r < 8; ++r) {
        const float inv = 1.0f / lsum[r];
        const int row = r + (hi ? 8 : 0);
#pragma unroll
        for (int t = 0; t < 4; ++t)
            Ob[(size_t)row * D_MODEL + t * 16 + laneLo] = (bf16_t)(o[t].f[r] * inv);
    }
}

// ---------------------------------------------------------------------------
extern "C" void kernel_launch(void* const* d_in, const int* in_sizes, int n_in,
                              void* d_out, int out_size, void* d_ws, size_t ws_size,
                              hipStream_t stream) {
    const float* Xdec = (const float*)d_in[0];
    const float* Xenc = (const float*)d_in[1];
    const int*   mask = (const int*)d_in[2];
    const float* Wq = (const float*)d_in[3];  const float* bq = (const float*)d_in[4];
    const float* Wk = (const float*)d_in[5];  const float* bk = (const float*)d_in[6];
    const float* Wv = (const float*)d_in[7];  const float* bv = (const float*)d_in[8];
    const float* Wo = (const float*)d_in[9];  const float* bo = (const float*)d_in[10];
    float* out = (float*)d_out;

    char* ws = (char*)d_ws;
    const size_t MB = 1024 * 1024;
    bf16_t* Xdec_bf = (bf16_t*)(ws + 0);        //  8 MB: [4096,1024]
    bf16_t* Xenc_bf = (bf16_t*)(ws + 8  * MB);  // 16 MB: [8192,1024]
    bf16_t* Wq_bf   = (bf16_t*)(ws + 24 * MB);  //  2 MB
    bf16_t* Wk_bf   = (bf16_t*)(ws + 26 * MB);  //  2 MB
    bf16_t* Wv_bf   = (bf16_t*)(ws + 28 * MB);  //  2 MB
    bf16_t* Wo_bf   = (bf16_t*)(ws + 30 * MB);  //  2 MB
    bf16_t* Qw      = (bf16_t*)(ws + 32 * MB);  //  8 MB: [4096,1024]
    bf16_t* Kw      = (bf16_t*)(ws + 40 * MB);  // 16 MB: [8192,1024]
    bf16_t* Vtw     = (bf16_t*)(ws + 56 * MB);  // 16 MB: [B,H,64,2048]
    bf16_t* Ow      = (bf16_t*)(ws + 72 * MB);  //  8 MB: [4096,1024]

    const int Mdec = BATCH * LQ;     // 4096
    const int Menc = BATCH * LKV;    // 8192

    // Phase 0: fp32 -> bf16
    cvt_f32_bf16<<<(Mdec * D_MODEL) / 1024, 256, 0, stream>>>(Xdec, Xdec_bf, (Mdec * D_MODEL) / 4);
    cvt_f32_bf16<<<(Menc * D_MODEL) / 1024, 256, 0, stream>>>(Xenc, Xenc_bf, (Menc * D_MODEL) / 4);
    cvt_f32_bf16<<<(D_MODEL * D_MODEL) / 1024, 256, 0, stream>>>(Wq, Wq_bf, (D_MODEL * D_MODEL) / 4);
    cvt_f32_bf16<<<(D_MODEL * D_MODEL) / 1024, 256, 0, stream>>>(Wk, Wk_bf, (D_MODEL * D_MODEL) / 4);
    cvt_f32_bf16<<<(D_MODEL * D_MODEL) / 1024, 256, 0, stream>>>(Wv, Wv_bf, (D_MODEL * D_MODEL) / 4);
    cvt_f32_bf16<<<(D_MODEL * D_MODEL) / 1024, 256, 0, stream>>>(Wo, Wo_bf, (D_MODEL * D_MODEL) / 4);

    // Phase 1: projections (Q scaled by 1/sqrt(Dh)=0.125 folded in)
    dim3 gq(D_MODEL / 128, Mdec / 128), ge(D_MODEL / 128, Menc / 128);
    gemm_bf16_nt<<<gq, 256, 0, stream>>>(Xdec_bf, Wq_bf, bq, Qw,  Mdec, D_MODEL, D_MODEL, 0, 0.125f);
    gemm_bf16_nt<<<ge, 256, 0, stream>>>(Xenc_bf, Wk_bf, bk, Kw,  Menc, D_MODEL, D_MODEL, 0, 1.0f);
    gemm_bf16_nt<<<ge, 256, 0, stream>>>(Xenc_bf, Wv_bf, bv, Vtw, Menc, D_MODEL, D_MODEL, 2, 1.0f);

    // Phase 2: flash attention (512 WGs x 8 waves; WG = one (b,h), 128 q rows)
    attn_flash<<<BATCH * NHEAD * (LQ / 128), 256, 0, stream>>>(Qw, Kw, Vtw, mask, Ow);

    // Phase 3: output projection -> fp32 d_out
    gemm_bf16_nt<<<gq, 256, 0, stream>>>(Ow, Wo_bf, bo, out, Mdec, D_MODEL, D_MODEL, 1, 1.0f);
}
